// DecoderRNN_16441134809147
// MI455X (gfx1250) — compile-verified, matrix operand order checked by reading, not run on hardware
//
#include <hip/hip_runtime.h>
#include <hip/hip_bf16.h>

typedef __attribute__((ext_vector_type(16))) _Float16 v16h;
typedef __attribute__((ext_vector_type(8)))  _Float16 v8h;
typedef __attribute__((ext_vector_type(8)))  float    v8f;

#define DEV __device__ __forceinline__

constexpr int B = 32, P = 196, ENC = 2048, DEC = 512, EMB = 512, ATTD = 512,
              V = 32000, L = 21, T = 20;
constexpr int G4 = 4 * DEC;            // 2048 LSTM gate width
constexpr int HP = EMB + ENC;          // 2560  (x = [emb | gate*awe])
constexpr int XK = HP + DEC;           // 3072  (x_full = [x | h])

// ---------------- workspace layout (bytes) ----------------
constexpr size_t OFF_SORT   = 0;                                   // 32 int sort_ind + 32 int dec_lens
constexpr size_t OFF_MEAN16 = 256;                                 // B*ENC f16
constexpr size_t OFF_H      = OFF_MEAN16 + (size_t)B * ENC * 2;    // B*DEC f32
constexpr size_t OFF_C      = OFF_H      + (size_t)B * DEC * 4;
constexpr size_t OFF_ATT2   = OFF_C      + (size_t)B * DEC * 4;    // B*ATTD f32
constexpr size_t OFF_GATE   = OFF_ATT2   + (size_t)B * ATTD * 4;   // B*ENC f32
constexpr size_t OFF_G      = OFF_GATE   + (size_t)B * ENC * 4;    // B*G4 f32
constexpr size_t OFF_HF16   = OFF_G      + (size_t)B * G4 * 4;     // B*DEC f16
constexpr size_t OFF_HPRED  = OFF_HF16   + (size_t)B * DEC * 2;    // B*DEC f16 (masked h_new)
constexpr size_t OFF_X16    = OFF_HPRED  + (size_t)B * DEC * 2;    // B*XK  f16
constexpr size_t OFF_WEA16  = OFF_X16    + (size_t)B * XK * 2;     // ATTD*ENC f16
constexpr size_t OFF_WH0C0  = OFF_WEA16  + (size_t)ATTD * ENC * 2; // (2*DEC)*ENC f16  [W_h0; W_c0]
constexpr size_t OFF_WDAFB  = OFF_WH0C0  + (size_t)2 * DEC * ENC * 2; // HP*DEC f16 [W_da; W_fb]
constexpr size_t OFF_WCAT   = OFF_WDAFB  + (size_t)HP * DEC * 2;   // G4*XK f16 [W_ih | W_hh]
constexpr size_t OFF_WFC16  = OFF_WCAT   + (size_t)G4 * XK * 2;    // V*DEC f16
constexpr size_t OFF_ATT1   = OFF_WFC16  + (size_t)V * DEC * 2;    // B*P*ATTD f32
constexpr size_t OFF_ENC16  = OFF_ATT1   + (size_t)B * P * ATTD * 4; // B*P*ENC f16 (sorted)

// ---------------- WMMA helpers ----------------
// 16-bit A-matrix 16x32 tile layout (ISA 7.12.2): lane l holds row l&15,
// K-halves selected by l>>4: Ks {koff..koff+7} and {koff+16..koff+23},
// koff = (l>>4)*8. Two 16-byte loads per lane.  The B operand of X @ W^T
// (W row-major N x K) loads W's rows with the same layout (B^T rows = W rows).
DEV v16h ld_tile(const _Float16* __restrict__ p, int ld) {
    const int lane = threadIdx.x & 31;
    const _Float16* q = p + (lane & 15) * ld + ((lane >> 4) << 3);
    union { v16h v; v8h h[2]; } u;
    u.h[0] = *(const v8h*)(q);
    u.h[1] = *(const v8h*)(q + 16);
    return u.v;
}

DEV v8f wmma16(v16h a, v16h b, v8f c) {
    return __builtin_amdgcn_wmma_f32_16x16x32_f16(false, a, false, b, (short)0, c,
                                                  false, false);
}

DEV float sigmoidf(float x) { return 1.f / (1.f + __expf(-x)); }

// ---------------- prep kernels ----------------
__global__ void sort_kernel(const int* __restrict__ caplens,
                            int* __restrict__ sort_ind, int* __restrict__ dec_lens) {
    int i = threadIdx.x;                 // 0..31
    int li = caplens[i];
    int rank = 0;                        // stable argsort of -lens
    for (int j = 0; j < B; ++j) {
        int lj = caplens[j];
        if (lj > li || (lj == li && j < i)) ++rank;
    }
    sort_ind[rank] = i;
    dec_lens[rank] = li - 1;
}

__global__ void gather_enc_f16(const float* __restrict__ enc,
                               const int* __restrict__ sort_ind,
                               _Float16* __restrict__ out) {
    long idx = (long)blockIdx.x * blockDim.x + threadIdx.x;
    if (idx >= (long)B * P * ENC) return;
    int b = (int)(idx / ((long)P * ENC));
    long rem = idx % ((long)P * ENC);
    out[idx] = (_Float16)enc[(long)sort_ind[b] * P * ENC + rem];
}

__global__ void cvt_f16(const float* __restrict__ src, _Float16* __restrict__ dst, long n) {
    long i = (long)blockIdx.x * blockDim.x + threadIdx.x;
    if (i < n) dst[i] = (_Float16)src[i];
}

// column concat: dst (G4 x XK) = [W_ih (G4 x HP) | W_hh (G4 x DEC)]
__global__ void build_wcat(const float* __restrict__ W_ih, const float* __restrict__ W_hh,
                           _Float16* __restrict__ dst) {
    long i = (long)blockIdx.x * blockDim.x + threadIdx.x;
    if (i >= (long)G4 * XK) return;
    int g = (int)(i / XK), j = (int)(i % XK);
    float v = (j < HP) ? W_ih[(long)g * HP + j] : W_hh[(long)g * DEC + (j - HP)];
    dst[i] = (_Float16)v;
}

// row concat: dst ((rowsA+rowsB) x k) = [A; Bm]
__global__ void build_rowcat(const float* __restrict__ A, int rowsA,
                             const float* __restrict__ Bm, int rowsB, int k,
                             _Float16* __restrict__ dst) {
    long i = (long)blockIdx.x * blockDim.x + threadIdx.x;
    if (i >= (long)(rowsA + rowsB) * k) return;
    int r = (int)(i / k), j = (int)(i % k);
    float v = (r < rowsA) ? A[(long)r * k + j] : Bm[(long)(r - rowsA) * k + j];
    dst[i] = (_Float16)v;
}

__global__ void mean_enc_f16(const float* __restrict__ enc,
                             const int* __restrict__ sort_ind,
                             _Float16* __restrict__ mean16) {
    int b = blockIdx.x;
    const float* src = enc + (long)sort_ind[b] * P * ENC;
    for (int e = threadIdx.x; e < ENC; e += blockDim.x) {
        float s = 0.f;
        for (int p = 0; p < P; ++p) s += src[(long)p * ENC + e];
        mean16[b * ENC + e] = (_Float16)(s * (1.f / (float)P));
    }
}

// ---------------- one-time WMMA GEMMs ----------------
// h0/c0 : mean16(32x2048) @ [W_h0;W_c0]^T -> 32x1024.  64 waves, 8 blocks x 256.
__global__ void h0c0_gemm(const _Float16* __restrict__ mean16, const _Float16* __restrict__ w,
                          const float* __restrict__ b_h0, const float* __restrict__ b_c0,
                          float* __restrict__ h, float* __restrict__ c,
                          _Float16* __restrict__ hf16, _Float16* __restrict__ x16) {
    int wave = (int)((blockIdx.x * blockDim.x + threadIdx.x) >> 5);  // 0..63
    int n0 = wave * 16;
    const _Float16* bp = w + (long)n0 * ENC;
    v8f c0 = {}, c1 = {};
    for (int k = 0; k < ENC; k += 32) {
        v16h bt = ld_tile(bp + k, ENC);
        c0 = wmma16(ld_tile(mean16 + k, ENC), bt, c0);
        c1 = wmma16(ld_tile(mean16 + 16 * ENC + k, ENC), bt, c1);
    }
    int lane = threadIdx.x & 31, col = lane & 15, rb = (lane >> 4) * 8;
    int cg = n0 + col;
    for (int i = 0; i < 8; ++i) {
        int r0 = rb + i, r1 = 16 + rb + i;
        if (cg < DEC) {
            float bv = b_h0[cg];
            float h0v = c0[i] + bv, h1v = c1[i] + bv;
            h[r0 * DEC + cg] = h0v;               h[r1 * DEC + cg] = h1v;
            hf16[r0 * DEC + cg] = (_Float16)h0v;  hf16[r1 * DEC + cg] = (_Float16)h1v;
            x16[(long)r0 * XK + HP + cg] = (_Float16)h0v;
            x16[(long)r1 * XK + HP + cg] = (_Float16)h1v;
        } else {
            int cc = cg - DEC;
            float bv = b_c0[cc];
            c[r0 * DEC + cc] = c0[i] + bv;
            c[r1 * DEC + cc] = c1[i] + bv;
        }
    }
}

// att1 : enc16(6272x2048) @ W_ea^T(2048x512) -> 6272x512.  12544 tiles, 1568 blocks x 256.
__global__ void att1_gemm(const _Float16* __restrict__ enc16, const _Float16* __restrict__ wea16,
                          const float* __restrict__ b_ea, float* __restrict__ att1) {
    int wid = (int)((blockIdx.x * blockDim.x + threadIdx.x) >> 5);
    int mt = wid >> 5, nt = wid & 31;
    const _Float16* ap = enc16 + (long)mt * 16 * ENC;
    const _Float16* bp = wea16 + (long)nt * 16 * ENC;
    v8f acc = {};
    for (int k = 0; k < ENC; k += 32)
        acc = wmma16(ld_tile(ap + k, ENC), ld_tile(bp + k, ENC), acc);
    int lane = threadIdx.x & 31, col = lane & 15, rb = (lane >> 4) * 8;
    int cg = nt * 16 + col;
    float bias = b_ea[cg];
    for (int i = 0; i < 8; ++i) {
        int r = mt * 16 + rb + i;
        att1[(long)r * ATTD + cg] = acc[i] + bias;
    }
}

// ---------------- per-step kernels ----------------
// h(32x512) @ [W_da;W_fb]^T -> att2(32x512), gate=sigmoid(...)(32x2048). 160 waves.
__global__ void hproj_gemm(const _Float16* __restrict__ hf16, const _Float16* __restrict__ wdafb,
                           const float* __restrict__ b_da, const float* __restrict__ b_fb,
                           float* __restrict__ att2, float* __restrict__ gate) {
    int wave = (int)((blockIdx.x * blockDim.x + threadIdx.x) >> 5);  // 0..159
    int n0 = wave * 16;
    const _Float16* bp = wdafb + (long)n0 * DEC;
    v8f c0 = {}, c1 = {};
    for (int k = 0; k < DEC; k += 32) {
        v16h bt = ld_tile(bp + k, DEC);
        c0 = wmma16(ld_tile(hf16 + k, DEC), bt, c0);
        c1 = wmma16(ld_tile(hf16 + 16 * DEC + k, DEC), bt, c1);
    }
    int lane = threadIdx.x & 31, col = lane & 15, rb = (lane >> 4) * 8;
    int cg = n0 + col;                      // 512 boundary is tile-aligned: uniform branch
    for (int i = 0; i < 8; ++i) {
        int r0 = rb + i, r1 = 16 + rb + i;
        if (cg < ATTD) {
            float bv = b_da[cg];
            att2[r0 * ATTD + cg] = c0[i] + bv;
            att2[r1 * ATTD + cg] = c1[i] + bv;
        } else {
            int cc = cg - ATTD;
            float bv = b_fb[cc];
            gate[r0 * ENC + cc] = sigmoidf(c0[i] + bv);
            gate[r1 * ENC + cc] = sigmoidf(c1[i] + bv);
        }
    }
}

// attention scores + softmax + awe + x assembly.  One block (256 thr) per batch row.
__global__ void attn_step(const float* __restrict__ enc, const int* __restrict__ sort_ind,
                          const int* __restrict__ dec_lens,
                          const float* __restrict__ att1, const float* __restrict__ att2,
                          const float* __restrict__ w_fa, const float* __restrict__ b_fa,
                          const float* __restrict__ gate,
                          const int* __restrict__ caps, const float* __restrict__ emb_W,
                          _Float16* __restrict__ x16, float* __restrict__ alphas_out, int t) {
    __shared__ float att2s[ATTD];
    __shared__ float wfas[ATTD];
    __shared__ float ebuf[P];
    __shared__ float red[2];
    int b = blockIdx.x, tid = threadIdx.x;
    for (int k = tid; k < ATTD; k += blockDim.x) {
        att2s[k] = att2[b * ATTD + k];
        wfas[k]  = w_fa[k];
    }
    __syncthreads();
    int lane = tid & 31, wave = tid >> 5;          // 8 waves
    const float* a1 = att1 + (long)b * P * ATTD;
    for (int p = wave; p < P; p += 8) {
        const float* row = a1 + (long)p * ATTD;
        float s = 0.f;
        for (int k = lane; k < ATTD; k += 32)
            s += fmaxf(row[k] + att2s[k], 0.f) * wfas[k];
        for (int m = 16; m > 0; m >>= 1) s += __shfl_xor(s, m, 32);
        if (lane == 0) ebuf[p] = s + b_fa[0];
    }
    __syncthreads();
    if (tid == 0) {
        float mx = ebuf[0];
        for (int p = 1; p < P; ++p) mx = fmaxf(mx, ebuf[p]);
        red[0] = mx;
    }
    __syncthreads();
    if (tid < P) ebuf[tid] = __expf(ebuf[tid] - red[0]);
    __syncthreads();
    if (tid == 0) {
        float s = 0.f;
        for (int p = 0; p < P; ++p) s += ebuf[p];
        red[1] = 1.f / s;
    }
    __syncthreads();
    float m = (t < dec_lens[b]) ? 1.f : 0.f;
    if (tid < P) {
        float al = ebuf[tid] * red[1];
        ebuf[tid] = al;
        alphas_out[((long)b * T + t) * P + tid] = al * m;
    }
    __syncthreads();
    const float* encb = enc + (long)sort_ind[b] * P * ENC;
    for (int e = tid; e < ENC; e += blockDim.x) {
        float s = 0.f;
        for (int p = 0; p < P; ++p) s += ebuf[p] * encb[(long)p * ENC + e];
        x16[(long)b * XK + EMB + e] = (_Float16)(gate[b * ENC + e] * s);
    }
    int cap = caps[sort_ind[b] * L + t];
    const float* er = emb_W + (long)cap * EMB;
    for (int j = tid; j < EMB; j += blockDim.x)
        x16[(long)b * XK + j] = (_Float16)er[j];
}

// g = x_full(32x3072) @ [W_ih|W_hh]^T -> 32x2048.  128 waves, 16 blocks x 256.
__global__ void lstm_gemm(const _Float16* __restrict__ x16, const _Float16* __restrict__ wcat,
                          const float* __restrict__ b_ih, const float* __restrict__ b_hh,
                          float* __restrict__ g) {
    int wave = (int)((blockIdx.x * blockDim.x + threadIdx.x) >> 5);  // 0..127
    int n0 = wave * 16;
    const _Float16* bp = wcat + (long)n0 * XK;
    v8f c0 = {}, c1 = {};
    for (int k = 0; k < XK; k += 32) {
        v16h bt = ld_tile(bp + k, XK);
        c0 = wmma16(ld_tile(x16 + k, XK), bt, c0);
        c1 = wmma16(ld_tile(x16 + 16 * XK + k, XK), bt, c1);
    }
    int lane = threadIdx.x & 31, col = lane & 15, rb = (lane >> 4) * 8;
    int cg = n0 + col;
    float bias = b_ih[cg] + b_hh[cg];
    for (int i = 0; i < 8; ++i) {
        g[(rb + i) * G4 + cg]      = c0[i] + bias;
        g[(16 + rb + i) * G4 + cg] = c1[i] + bias;
    }
}

__global__ void lstm_update(const float* __restrict__ g, const int* __restrict__ dec_lens,
                            float* __restrict__ h, float* __restrict__ c,
                            _Float16* __restrict__ hf16, _Float16* __restrict__ hpred,
                            _Float16* __restrict__ x16, int t) {
    int idx = blockIdx.x * blockDim.x + threadIdx.x;
    if (idx >= B * DEC) return;
    int b = idx / DEC, v = idx % DEC;
    const float* gr = g + (long)b * G4;
    float si = sigmoidf(gr[v]);
    float sf = sigmoidf(gr[DEC + v]);
    float gg = gr[2 * DEC + v];
    float so = sigmoidf(gr[3 * DEC + v]);
    float cn = sf * c[idx] + si * tanhf(gg);
    float hn = so * tanhf(cn);
    bool m = t < dec_lens[b];
    hpred[idx] = (_Float16)(m ? hn : 0.f);
    float hq = m ? hn : h[idx];
    float cq = m ? cn : c[idx];
    h[idx] = hq;
    c[idx] = cq;
    hf16[idx] = (_Float16)hq;
    x16[(long)b * XK + HP + v] = (_Float16)hq;
}

// preds = h_new(32x512) @ W_fc^T(512x32000) + b_fc, masked.  2000 waves, 500 blocks x 128.
__global__ void preds_gemm(const _Float16* __restrict__ hpred, const _Float16* __restrict__ wfc16,
                           const float* __restrict__ b_fc, const int* __restrict__ dec_lens,
                           float* __restrict__ preds_out, int t) {
    int wave = (int)((blockIdx.x * blockDim.x + threadIdx.x) >> 5);  // 0..1999
    int n0 = wave * 16;
    const _Float16* bp = wfc16 + (long)n0 * DEC;
    v8f c0 = {}, c1 = {};
    for (int k = 0; k < DEC; k += 32) {
        v16h bt = ld_tile(bp + k, DEC);
        c0 = wmma16(ld_tile(hpred + k, DEC), bt, c0);
        c1 = wmma16(ld_tile(hpred + 16 * DEC + k, DEC), bt, c1);
    }
    int lane = threadIdx.x & 31, col = lane & 15, rb = (lane >> 4) * 8;
    int cg = n0 + col;
    float bias = b_fc[cg];
    for (int i = 0; i < 8; ++i) {
        int r0 = rb + i, r1 = 16 + rb + i;
        float m0 = (t < dec_lens[r0]) ? 1.f : 0.f;
        float m1 = (t < dec_lens[r1]) ? 1.f : 0.f;
        preds_out[((long)r0 * T + t) * V + cg] = (c0[i] + bias) * m0;
        preds_out[((long)r1 * T + t) * V + cg] = (c1[i] + bias) * m1;
    }
}

// ---------------- host launcher ----------------
extern "C" void kernel_launch(void* const* d_in, const int* in_sizes, int n_in,
                              void* d_out, int out_size, void* d_ws, size_t ws_size,
                              hipStream_t stream) {
    const float* enc     = (const float*)d_in[0];
    const int*   caps    = (const int*)  d_in[1];
    const int*   caplens = (const int*)  d_in[2];
    const float* emb_W   = (const float*)d_in[3];
    const float* W_ea    = (const float*)d_in[4];
    const float* b_ea    = (const float*)d_in[5];
    const float* W_da    = (const float*)d_in[6];
    const float* b_da    = (const float*)d_in[7];
    const float* w_fa    = (const float*)d_in[8];
    const float* b_fa    = (const float*)d_in[9];
    const float* W_ih    = (const float*)d_in[10];
    const float* b_ih    = (const float*)d_in[11];
    const float* W_hh    = (const float*)d_in[12];
    const float* b_hh    = (const float*)d_in[13];
    const float* W_h0    = (const float*)d_in[14];
    const float* b_h0    = (const float*)d_in[15];
    const float* W_c0    = (const float*)d_in[16];
    const float* b_c0    = (const float*)d_in[17];
    const float* W_fb    = (const float*)d_in[18];
    const float* b_fb    = (const float*)d_in[19];
    const float* W_fc    = (const float*)d_in[20];
    const float* b_fc    = (const float*)d_in[21];

    char* ws = (char*)d_ws;
    int*      sort_ind = (int*)(ws + OFF_SORT);
    int*      dec_lens = sort_ind + 32;
    _Float16* mean16   = (_Float16*)(ws + OFF_MEAN16);
    float*    h        = (float*)(ws + OFF_H);
    float*    c        = (float*)(ws + OFF_C);
    float*    att2     = (float*)(ws + OFF_ATT2);
    float*    gate     = (float*)(ws + OFF_GATE);
    float*    g        = (float*)(ws + OFF_G);
    _Float16* hf16     = (_Float16*)(ws + OFF_HF16);
    _Float16* hpred    = (_Float16*)(ws + OFF_HPRED);
    _Float16* x16      = (_Float16*)(ws + OFF_X16);
    _Float16* wea16    = (_Float16*)(ws + OFF_WEA16);
    _Float16* wh0c0    = (_Float16*)(ws + OFF_WH0C0);
    _Float16* wdafb    = (_Float16*)(ws + OFF_WDAFB);
    _Float16* wcat     = (_Float16*)(ws + OFF_WCAT);
    _Float16* wfc16    = (_Float16*)(ws + OFF_WFC16);
    float*    att1     = (float*)(ws + OFF_ATT1);
    _Float16* enc16    = (_Float16*)(ws + OFF_ENC16);

    float* preds_out  = (float*)d_out;
    float* alphas_out = preds_out + (long)B * T * V;

    auto nb = [](long n) { return (unsigned)((n + 255) / 256); };

    // -------- prep --------
    sort_kernel<<<1, 32, 0, stream>>>(caplens, sort_ind, dec_lens);
    gather_enc_f16<<<nb((long)B * P * ENC), 256, 0, stream>>>(enc, sort_ind, enc16);
    cvt_f16<<<nb((long)ATTD * ENC), 256, 0, stream>>>(W_ea, wea16, (long)ATTD * ENC);
    cvt_f16<<<nb((long)V * DEC), 256, 0, stream>>>(W_fc, wfc16, (long)V * DEC);
    build_rowcat<<<nb((long)2 * DEC * ENC), 256, 0, stream>>>(W_h0, DEC, W_c0, DEC, ENC, wh0c0);
    build_rowcat<<<nb((long)HP * DEC), 256, 0, stream>>>(W_da, ATTD, W_fb, ENC, DEC, wdafb);
    build_wcat<<<nb((long)G4 * XK), 256, 0, stream>>>(W_ih, W_hh, wcat);
    mean_enc_f16<<<B, 256, 0, stream>>>(enc, sort_ind, mean16);

    // -------- one-time WMMA GEMMs --------
    h0c0_gemm<<<8, 256, 0, stream>>>(mean16, wh0c0, b_h0, b_c0, h, c, hf16, x16);
    att1_gemm<<<(B * P / 16) * (ATTD / 16) / 8, 256, 0, stream>>>(enc16, wea16, b_ea, att1);

    // -------- sequential decode --------
    for (int t = 0; t < T; ++t) {
        hproj_gemm<<<HP / 16 / 8, 256, 0, stream>>>(hf16, wdafb, b_da, b_fb, att2, gate);
        attn_step<<<B, 256, 0, stream>>>(enc, sort_ind, dec_lens, att1, att2, w_fa, b_fa,
                                         gate, caps, emb_W, x16, alphas_out, t);
        lstm_gemm<<<G4 / 16 / 8, 256, 0, stream>>>(x16, wcat, b_ih, b_hh, g);
        lstm_update<<<nb((long)B * DEC), 256, 0, stream>>>(g, dec_lens, h, c, hf16, hpred,
                                                           x16, t);
        preds_gemm<<<(V / 16) / 4, 128, 0, stream>>>(hpred, wfc16, b_fc, dec_lens,
                                                     preds_out, t);
    }
}
